// DemoTransformer_37778532336013
// MI455X (gfx1250) — matrix operand order repaired and checked by reading, not compiled, not run on hardware
//
#include <hip/hip_runtime.h>
#include <math.h>
#include <stdint.h>

// ---------------- config ----------------
#define SEQ    2048
#define DMODEL 1024
#define DMLP_  4096
#define NHEAD  16
#define DHEAD  64
#define VOCAB  50257
#define NLAYER 2
#define EPS_   1e-5f
#define IGNORE_ (-1e5f)

// ---------------- types ----------------
typedef __attribute__((ext_vector_type(16))) __bf16 v16bf;
typedef __attribute__((ext_vector_type(8)))  float  v8f;
typedef __attribute__((ext_vector_type(8)))  unsigned short us8;

union AFrag { v16bf v; unsigned short s[16]; };

__device__ __forceinline__ unsigned short f2bf(float f) {
    union { float f; unsigned u; } c; c.f = f;
    unsigned u = c.u;
    u += 0x7FFFu + ((u >> 16) & 1u);          // round-to-nearest-even
    return (unsigned short)(u >> 16);
}

// CDNA5 async global->LDS copy, 16B per lane, tracked by ASYNCcnt.
__device__ __forceinline__ void async_ld_b128(const unsigned short* lds_dst, const unsigned short* gsrc) {
    unsigned lds_off = (unsigned)(uintptr_t)lds_dst;   // low 32 bits of generic LDS ptr = LDS offset
    asm volatile("global_load_async_to_lds_b128 %0, %1, off"
                 :: "v"(lds_off), "v"(gsrc) : "memory");
}

// ---------------- elementwise kernels ----------------
__global__ void k_embed(const int* __restrict__ tok, const float* __restrict__ WE,
                        const float* __restrict__ Wpos, float* __restrict__ resid) {
    size_t idx = (size_t)blockIdx.x * blockDim.x + threadIdx.x;
    if (idx >= (size_t)SEQ * DMODEL) return;
    int s = (int)(idx >> 10);
    int d = (int)(idx & 1023);
    resid[idx] = WE[(size_t)tok[s] * DMODEL + d] + Wpos[idx];
}

// f32 [K x N] row-major -> bf16 [N x K] (n-major, k-contiguous)
__global__ void k_conv_T(const float* __restrict__ in, unsigned short* __restrict__ out,
                         int K, int N) {
    size_t idx = (size_t)blockIdx.x * blockDim.x + threadIdx.x;
    if (idx >= (size_t)K * N) return;
    int n = (int)(idx / K);
    int k = (int)(idx % K);
    out[idx] = f2bf(in[(size_t)k * N + n]);
}

// W_Q[h, d, e] (H x DM x DH) -> bf16 [(h*64+e) x DM] (n-major, k(=d)-contiguous)
__global__ void k_conv_qkvwT(const float* __restrict__ in, unsigned short* __restrict__ out) {
    size_t idx = (size_t)blockIdx.x * blockDim.x + threadIdx.x;
    if (idx >= (size_t)DMODEL * NHEAD * DHEAD) return;
    int n = (int)(idx >> 10);          // h*64+e
    int d = (int)(idx & 1023);
    int h = n >> 6, e = n & 63;
    out[idx] = f2bf(in[(size_t)h * DMODEL * DHEAD + (size_t)d * DHEAD + e]);
}

__global__ void k_layernorm_bf16(const float* __restrict__ X, const float* __restrict__ w,
                                 const float* __restrict__ b, unsigned short* __restrict__ out) {
    __shared__ float r1[256], r2[256];
    int row = blockIdx.x, tid = threadIdx.x;
    const float* x = X + (size_t)row * DMODEL;
    float s = 0.f, sq = 0.f;
    for (int j = tid; j < DMODEL; j += 256) { float v = x[j]; s += v; sq += v * v; }
    r1[tid] = s; r2[tid] = sq; __syncthreads();
    for (int k = 128; k > 0; k >>= 1) {
        if (tid < k) { r1[tid] += r1[tid + k]; r2[tid] += r2[tid + k]; }
        __syncthreads();
    }
    float mean = r1[0] / DMODEL;
    float var  = r2[0] / DMODEL - mean * mean;
    float rstd = rsqrtf(var + EPS_);
    for (int j = tid; j < DMODEL; j += 256) {
        float v = (x[j] - mean) * rstd * w[j] + b[j];
        out[(size_t)row * DMODEL + j] = f2bf(v);
    }
}

__global__ void k_softmax_causal_bf16(const float* __restrict__ S, unsigned short* __restrict__ P,
                                      float scale) {
    __shared__ float red[256];
    int row = blockIdx.x, tid = threadIdx.x;
    const float* sr = S + (size_t)row * SEQ;
    float mx = -3.4e38f;
    for (int j = tid; j < SEQ; j += 256) {
        float v = (j > row) ? IGNORE_ : sr[j] * scale;
        mx = fmaxf(mx, v);
    }
    red[tid] = mx; __syncthreads();
    for (int k = 128; k > 0; k >>= 1) { if (tid < k) red[tid] = fmaxf(red[tid], red[tid + k]); __syncthreads(); }
    mx = red[0]; __syncthreads();
    float sum = 0.f;
    for (int j = tid; j < SEQ; j += 256) {
        float v = (j > row) ? IGNORE_ : sr[j] * scale;
        sum += expf(v - mx);
    }
    red[tid] = sum; __syncthreads();
    for (int k = 128; k > 0; k >>= 1) { if (tid < k) red[tid] += red[tid + k]; __syncthreads(); }
    float inv = 1.f / red[0];
    for (int j = tid; j < SEQ; j += 256) {
        float v = (j > row) ? IGNORE_ : sr[j] * scale;
        P[(size_t)row * SEQ + j] = f2bf(expf(v - mx) * inv);
    }
}

__global__ void k_gelu(const float* __restrict__ pre, float* __restrict__ outf,
                       unsigned short* __restrict__ outbf, size_t n) {
    size_t idx = (size_t)blockIdx.x * blockDim.x + threadIdx.x;
    if (idx >= n) return;
    float x = pre[idx];
    float g = 0.5f * x * (1.0f + erff(x * 0.70710678118654752f));  // exact GELU
    if (outf) outf[idx] = g;
    outbf[idx] = f2bf(g);
}

// ---------------- WMMA bf16 GEMM (async double-buffered, templated tiling) ----------------
// C[M,N] = A[M,K] * B^T where A row-major (lda), B n-major/k-contiguous (ldb = k-stride).
// (+bias[n]) (+resid[m*ldr+n]); f32 out Cf[ldc]; bf16 out Cbf[ldc] (transC: Cbf[col*ldc+row]).
// Requirements: K%32==0; M%BM==0; lda,ldb multiples of 8; pointers 16B-aligned.
// N-edge blocks fall back to a synchronous scalar staging path.
template<int WM, int WN, int TM, int TN>
__global__ void __launch_bounds__(256)
k_gemm_bf16_wmma(const unsigned short* __restrict__ A, int lda,
                 const unsigned short* __restrict__ B, int ldb,
                 float* __restrict__ Cf, unsigned short* __restrict__ Cbf, int ldc, int transC,
                 const float* __restrict__ bias, const float* __restrict__ resid, int ldr,
                 int M, int N, int K) {
    constexpr int BM = WM * TM * 16;
    constexpr int BN = WN * TN * 16;
    __shared__ unsigned short As[2][BM * 32];   // (m x k), k contiguous
    __shared__ unsigned short Bs[2][BN * 32];   // (n x k), k contiguous

    const int tid  = threadIdx.x;
    const int lane = tid & 31;
    const int wave = tid >> 5;              // 0..7
    const int wm   = wave / WN;
    const int wn   = wave % WN;
    const int half = lane >> 4;             // 0/1
    const int l16  = lane & 15;

    const int blockM0 = blockIdx.y * BM;
    const int blockN0 = blockIdx.x * BN;
    const bool nedge  = (blockN0 + BN > N);

    v8f acc[TM][TN];
    #pragma unroll
    for (int t = 0; t < TM; ++t)
        #pragma unroll
        for (int u = 0; u < TN; ++u) acc[t][u] = (v8f){};

    // fragment build + 4 WMMAs from buffer `buf`
    auto compute = [&](int buf) {
        AFrag af[TM], bf[TN];
        #pragma unroll
        for (int t = 0; t < TM; ++t) {
            const unsigned short* ap = &As[buf][((wm * TM + t) * 16 + l16) * 32];
            #pragma unroll
            for (int j = 0; j < 16; ++j) af[t].s[j] = ap[16 * half + j];
        }
        #pragma unroll
        for (int u = 0; u < TN; ++u) {
            const unsigned short* bp = &Bs[buf][((wn * TN + u) * 16 + l16) * 32 + 16 * half];
            #pragma unroll
            for (int j = 0; j < 16; ++j) bf[u].s[j] = bp[j];
        }
        #pragma unroll
        for (int t = 0; t < TM; ++t)
            #pragma unroll
            for (int u = 0; u < TN; ++u)
                acc[t][u] = __builtin_amdgcn_wmma_f32_16x16x32_bf16(
                                false, af[t].v, false, bf[u].v, (short)0, acc[t][u], false, false);
    };

    if (!nedge) {
        // fully in-bounds: async global->LDS copies, double-buffered.
        // Per thread per tile: BM*4/256 + BN*4/256 == 3 async ops for both configs.
        auto issue = [&](int buf, int k0) {
            #pragma unroll
            for (int i = tid; i < BM * 4; i += 256) {
                int r = i >> 2, kk8 = (i & 3) * 8;
                async_ld_b128(&As[buf][r * 32 + kk8],
                              &A[(size_t)(blockM0 + r) * lda + k0 + kk8]);
            }
            #pragma unroll
            for (int i = tid; i < BN * 4; i += 256) {
                int nn = i >> 2, kk8 = (i & 3) * 8;
                async_ld_b128(&Bs[buf][nn * 32 + kk8],
                              &B[(size_t)(blockN0 + nn) * ldb + k0 + kk8]);
            }
        };
        issue(0, 0);
        int buf = 0;
        for (int k0 = 0; k0 < K; k0 += 32) {
            const bool more = (k0 + 32) < K;
            if (more) {
                issue(buf ^ 1, k0 + 32);
                asm volatile("s_wait_asynccnt 0x3" ::: "memory");  // current tile's 3 ops done
            } else {
                asm volatile("s_wait_asynccnt 0x0" ::: "memory");
            }
            __syncthreads();
            compute(buf);
            __syncthreads();
            buf ^= 1;
        }
    } else {
        // N-edge: synchronous staging with guards (single buffer)
        for (int k0 = 0; k0 < K; k0 += 32) {
            __syncthreads();
            #pragma unroll
            for (int i = tid; i < BM * 4; i += 256) {
                int r = i >> 2, kk8 = (i & 3) * 8;
                *(us8*)&As[0][r * 32 + kk8] =
                    *(const us8*)&A[(size_t)(blockM0 + r) * lda + k0 + kk8];
            }
            for (int i = tid; i < BN * 32; i += 256) {
                int nn = i >> 5, kk = i & 31;
                int gn = blockN0 + nn;
                Bs[0][nn * 32 + kk] = (gn < N) ? B[(size_t)gn * ldb + (k0 + kk)]
                                               : (unsigned short)0;
            }
            __syncthreads();
            compute(0);
        }
    }

    // ---- epilogue ----
    #pragma unroll
    for (int u = 0; u < TN; ++u) {
        int col = blockN0 + (wn * TN + u) * 16 + l16;
        if (col >= N) continue;
        float bv = bias ? bias[col] : 0.f;
        #pragma unroll
        for (int t = 0; t < TM; ++t) {
            #pragma unroll
            for (int rr = 0; rr < 8; ++rr) {
                int row = blockM0 + (wm * TM + t) * 16 + rr + 8 * half;
                float v = acc[t][u][rr] + bv;
                if (resid) v += resid[(size_t)row * ldr + col];
                if (Cf)  Cf[(size_t)row * ldc + col] = v;
                if (Cbf) {
                    if (transC) Cbf[(size_t)col * ldc + row] = f2bf(v);
                    else        Cbf[(size_t)row * ldc + col] = f2bf(v);
                }
            }
        }
    }
}

// ---------------- host orchestration ----------------
extern "C" void kernel_launch(void* const* d_in, const int* in_sizes, int n_in,
                              void* d_out, int out_size, void* d_ws, size_t ws_size,
                              hipStream_t stream) {
    (void)in_sizes; (void)n_in; (void)out_size; (void)ws_size;

    const int*   tokens = (const int*)  d_in[0];
    const float* W_E    = (const float*)d_in[1];
    const float* W_pos  = (const float*)d_in[2];
    const float* ln1_w  = (const float*)d_in[3];
    const float* ln1_b  = (const float*)d_in[4];
    const float* W_Q    = (const float*)d_in[5];
    const float* b_Q    = (const float*)d_in[6];
    const float* W_K    = (const float*)d_in[7];
    const float* b_K    = (const float*)d_in[8];
    const float* W_V    = (const float*)d_in[9];
    const float* b_V    = (const float*)d_in[10];
    const float* W_O    = (const float*)d_in[11];
    const float* b_O    = (const float*)d_in[12];
    const float* ln2_w  = (const float*)d_in[13];
    const float* ln2_b  = (const float*)d_in[14];
    const float* W_in   = (const float*)d_in[15];
    const float* b_in   = (const float*)d_in[16];
    const float* W_out  = (const float*)d_in[17];
    const float* b_out  = (const float*)d_in[18];
    const float* lnf_w  = (const float*)d_in[19];
    const float* lnf_b  = (const float*)d_in[20];
    const float* W_U    = (const float*)d_in[21];
    const float* b_U    = (const float*)d_in[22];

    float* out_logits = (float*)d_out;
    float* out_resid  = out_logits + (size_t)SEQ * VOCAB;
    float* out_post   = out_resid  + (size_t)SEQ * DMODEL;

    // workspace carve-up (256B aligned)
    char* base = (char*)d_ws;
    size_t off = 0;
    auto carve = [&](size_t bytes) -> char* {
        char* p = base + off;
        off = (off + bytes + 255) & ~(size_t)255;
        return p;
    };
    float*          resid   = (float*)         carve((size_t)SEQ * DMODEL * 4);
    unsigned short* xbf     = (unsigned short*)carve((size_t)SEQ * DMODEL * 2);
    unsigned short* qbf     = (unsigned short*)carve((size_t)SEQ * DMODEL * 2);
    unsigned short* kbf     = (unsigned short*)carve((size_t)SEQ * DMODEL * 2);
    unsigned short* vT      = (unsigned short*)carve((size_t)DMODEL * SEQ * 2);   // [h*64+e][s]
    unsigned short* zbf     = (unsigned short*)carve((size_t)SEQ * DMODEL * 2);
    float*          scores  = (float*)         carve((size_t)SEQ * SEQ * 4);
    unsigned short* pbf     = (unsigned short*)carve((size_t)SEQ * SEQ * 2);
    float*          pre     = (float*)         carve((size_t)SEQ * DMLP_ * 4);
    unsigned short* postbf  = (unsigned short*)carve((size_t)SEQ * DMLP_ * 2);
    unsigned short* wqT     = (unsigned short*)carve((size_t)DMODEL * DMODEL * 2);
    unsigned short* wkT     = (unsigned short*)carve((size_t)DMODEL * DMODEL * 2);
    unsigned short* wvT     = (unsigned short*)carve((size_t)DMODEL * DMODEL * 2);
    unsigned short* woT     = (unsigned short*)carve((size_t)DMODEL * DMODEL * 2);
    unsigned short* winT    = (unsigned short*)carve((size_t)DMODEL * DMLP_ * 2);
    unsigned short* woutT   = (unsigned short*)carve((size_t)DMLP_ * DMODEL * 2);
    unsigned short* wuT     = (unsigned short*)carve((size_t)VOCAB * DMODEL * 2);

    // wide tiles: 64x128 block, 8 waves of 32x32 (2x2 WMMA tiles each)
    auto gemmA = [&](const unsigned short* A, int lda, const unsigned short* B, int ldb,
                     float* Cf, unsigned short* Cbf, int ldc, int transC,
                     const float* bias, const float* res, int ldr, int M, int N, int K) {
        dim3 g((N + 127) / 128, (M + 63) / 64);
        k_gemm_bf16_wmma<2, 4, 2, 2><<<g, 256, 0, stream>>>(A, lda, B, ldb, Cf, Cbf, ldc, transC,
                                                            bias, res, ldr, M, N, K);
    };
    // narrow tiles: 128x64 block, 8 waves of 16x64 (1x4 WMMA tiles each) — for N=64 (P*V)
    auto gemmB = [&](const unsigned short* A, int lda, const unsigned short* B, int ldb,
                     float* Cf, unsigned short* Cbf, int ldc, int transC,
                     const float* bias, const float* res, int ldr, int M, int N, int K) {
        dim3 g((N + 63) / 64, (M + 127) / 128);
        k_gemm_bf16_wmma<8, 1, 1, 4><<<g, 256, 0, stream>>>(A, lda, B, ldb, Cf, Cbf, ldc, transC,
                                                            bias, res, ldr, M, N, K);
    };
    auto convT = [&](const float* in, unsigned short* out, int K, int N) {
        size_t n = (size_t)K * N;
        k_conv_T<<<(unsigned)((n + 255) / 256), 256, 0, stream>>>(in, out, K, N);
    };

    const float inv_sqrt_dh = 0.125f;  // 1/sqrt(64)

    // embedding
    {
        size_t n = (size_t)SEQ * DMODEL;
        k_embed<<<(unsigned)((n + 255) / 256), 256, 0, stream>>>(tokens, W_E, W_pos, resid);
    }
    // unembed weights once: [DMODEL x VOCAB] -> [VOCAB x DMODEL]
    convT(W_U, wuT, DMODEL, VOCAB);

    for (int l = 0; l < NLAYER; ++l) {
        const size_t wqkv_sz = (size_t)NHEAD * DMODEL * DHEAD;  // 1M
        {
            size_t n = (size_t)DMODEL * DMODEL;
            unsigned g = (unsigned)((n + 255) / 256);
            k_conv_qkvwT<<<g, 256, 0, stream>>>(W_Q + (size_t)l * wqkv_sz, wqT);
            k_conv_qkvwT<<<g, 256, 0, stream>>>(W_K + (size_t)l * wqkv_sz, wkT);
            k_conv_qkvwT<<<g, 256, 0, stream>>>(W_V + (size_t)l * wqkv_sz, wvT);
        }
        convT(W_O  + (size_t)l * DMODEL * DMODEL, woT,   DMODEL, DMODEL);
        convT(W_in + (size_t)l * DMODEL * DMLP_,  winT,  DMODEL, DMLP_);
        convT(W_out+ (size_t)l * DMLP_ * DMODEL,  woutT, DMLP_,  DMODEL);

        // LN1
        k_layernorm_bf16<<<SEQ, 256, 0, stream>>>(resid, ln1_w + l * DMODEL, ln1_b + l * DMODEL, xbf);

        // Q, K projections row-major bf16; V projection written transposed [h*64+e][s]
        gemmA(xbf, DMODEL, wqT, DMODEL, nullptr, qbf, DMODEL, 0,
              b_Q + (size_t)l * NHEAD * DHEAD, nullptr, 0, SEQ, DMODEL, DMODEL);
        gemmA(xbf, DMODEL, wkT, DMODEL, nullptr, kbf, DMODEL, 0,
              b_K + (size_t)l * NHEAD * DHEAD, nullptr, 0, SEQ, DMODEL, DMODEL);
        gemmA(xbf, DMODEL, wvT, DMODEL, nullptr, vT, SEQ, /*transC=*/1,
              b_V + (size_t)l * NHEAD * DHEAD, nullptr, 0, SEQ, DMODEL, DMODEL);

        // attention, head by head (buffers reused; stream-ordered)
        for (int h = 0; h < NHEAD; ++h) {
            // scores = Q_h * K_h^T   [SEQ x SEQ]  (B = kbf rows, k-contiguous)
            gemmA(qbf + h * DHEAD, DMODEL, kbf + h * DHEAD, DMODEL,
                  scores, nullptr, SEQ, 0, nullptr, nullptr, 0, SEQ, SEQ, DHEAD);
            // causal softmax with scale -> bf16 P
            k_softmax_causal_bf16<<<SEQ, 256, 0, stream>>>(scores, pbf, inv_sqrt_dh);
            // z_h = P * V_h   [SEQ x DHEAD]  (B = vT rows, k-contiguous)
            gemmB(pbf, SEQ, vT + (size_t)h * DHEAD * SEQ, SEQ,
                  nullptr, zbf + h * DHEAD, DMODEL, 0, nullptr, nullptr, 0, SEQ, DHEAD, SEQ);
        }

        // attn_out = z * W_O + b_O + resid   (in-place into resid)
        gemmA(zbf, DMODEL, woT, DMODEL, resid, nullptr, DMODEL, 0,
              b_O + (size_t)l * DMODEL, resid, DMODEL, SEQ, DMODEL, DMODEL);

        // LN2
        k_layernorm_bf16<<<SEQ, 256, 0, stream>>>(resid, ln2_w + l * DMODEL, ln2_b + l * DMODEL, xbf);

        // MLP in
        gemmA(xbf, DMODEL, winT, DMODEL, pre, nullptr, DMLP_, 0,
              b_in + (size_t)l * DMLP_, nullptr, 0, SEQ, DMLP_, DMODEL);

        // GELU (last layer also writes fp32 'post' output)
        {
            size_t n = (size_t)SEQ * DMLP_;
            k_gelu<<<(unsigned)((n + 255) / 256), 256, 0, stream>>>(
                pre, (l == NLAYER - 1) ? out_post : nullptr, postbf, n);
        }

        // MLP out + resid (in-place)
        gemmA(postbf, DMLP_, woutT, DMLP_, resid, nullptr, DMODEL, 0,
              b_out + (size_t)l * DMODEL, resid, DMODEL, SEQ, DMODEL, DMLP_);
    }

    // residual output
    hipMemcpyAsync(out_resid, resid, (size_t)SEQ * DMODEL * 4, hipMemcpyDeviceToDevice, stream);

    // final LN + unembed
    k_layernorm_bf16<<<SEQ, 256, 0, stream>>>(resid, lnf_w, lnf_b, xbf);
    gemmA(xbf, DMODEL, wuT, DMODEL, out_logits, nullptr, VOCAB, 0,
          b_U, nullptr, 0, SEQ, VOCAB, DMODEL);
}
